// CyCTransformer_66048007078343
// MI455X (gfx1250) — matrix operand order repaired and checked by reading, not compiled, not run on hardware
//
#include <hip/hip_runtime.h>
#include <math.h>

// ---------------- constants (match reference) ----------------
#define NLAYER 5
#define D_EMB  384
#define FF_DIM 1152
#define BATCH  4
#define NTOK   3600
#define SCALE_F 0.051031036307982884f   // 384^-0.5
#define LN_EPS  1e-5f

// ---------------- types for WMMA ----------------
typedef __attribute__((ext_vector_type(16))) __bf16  v16bf;
typedef __attribute__((ext_vector_type(8)))  float   v8f;
typedef __attribute__((ext_vector_type(4)))  unsigned int v4u;
typedef __attribute__((ext_vector_type(2)))  unsigned int v2u;
typedef __attribute__((ext_vector_type(4)))  float   v4f;

struct Pair16 { v4u lo, hi; };   // 32 bytes -> bit_cast to v16bf

static __device__ inline unsigned short f2bf_bits(float f) {
    unsigned u = __builtin_bit_cast(unsigned, f);
    unsigned r = u + 0x7FFFu + ((u >> 16) & 1u);   // round-to-nearest-even
    return (unsigned short)(r >> 16);
}
static __device__ inline unsigned pack2bf(float a, float b) {
    return (unsigned)f2bf_bits(a) | ((unsigned)f2bf_bits(b) << 16);
}

// ---------------------------------------------------------------------
// staging helpers: fp32 global -> bf16 LDS tiles
// A tile: 128 rows x 32 k   (lds_a: [m][k], 32 halves per row)
// B tile:  32 k   x 64 cols (lds_b: [k][n], 64 halves per row)
// Rows/cols are index-clamped (stores masked later); K-tail zero-fills A.
// ---------------------------------------------------------------------
template <bool FAST>
static __device__ inline void stage_A(unsigned short* __restrict__ lds_a,
                                      const float* __restrict__ A,
                                      int M, int K, int row0, int k0, int tid)
{
    const int mi = tid >> 3;           // 0..31
    const int ki = (tid & 7) << 2;     // 0,4,...,28
    #pragma unroll
    for (int r = 0; r < 4; ++r) {
        const int mm = mi + r * 32;
        int gm = row0 + mm; gm = (gm < M) ? gm : (M - 1);
        const float* ap = A + (size_t)gm * K + k0 + ki;
        float f0, f1, f2, f3;
        if (FAST) {
            v4f v = *reinterpret_cast<const v4f*>(ap);
            f0 = v.x; f1 = v.y; f2 = v.z; f3 = v.w;
            __builtin_prefetch(ap + 64, 0, 1);   // ~2 k-steps ahead
        } else {
            const int gk = k0 + ki;
            f0 = (gk     < K) ? ap[0] : 0.f;
            f1 = (gk + 1 < K) ? ap[1] : 0.f;
            f2 = (gk + 2 < K) ? ap[2] : 0.f;
            f3 = (gk + 3 < K) ? ap[3] : 0.f;
        }
        v2u packed; packed.x = pack2bf(f0, f1); packed.y = pack2bf(f2, f3);
        *reinterpret_cast<v2u*>(lds_a + mm * 32 + ki) = packed;
    }
}

template <int BT, bool FAST>
static __device__ inline void stage_B(unsigned short* __restrict__ lds_b,
                                      const float* __restrict__ Bm,
                                      int Ncols, int K, int col0, int k0, int tid)
{
    if (BT == 0) {
        // Bm: [K,Ncols]; k clamped (A zero-fill covers the tail), cols guarded
        const int kb = tid >> 3;           // 0..31
        const int nb = (tid & 7) << 3;     // 0,8,...,56
        int gk = k0 + kb; gk = (gk < K) ? gk : (K - 1);
        const float* bp = Bm + (size_t)gk * Ncols;
        const int gn = col0 + nb;
        float f[8];
        if (gn + 7 < Ncols) {
            v4f a = *reinterpret_cast<const v4f*>(bp + gn);
            v4f b = *reinterpret_cast<const v4f*>(bp + gn + 4);
            f[0]=a.x; f[1]=a.y; f[2]=a.z; f[3]=a.w;
            f[4]=b.x; f[5]=b.y; f[6]=b.z; f[7]=b.w;
        } else {
            #pragma unroll
            for (int j = 0; j < 8; ++j) {
                int n = gn + j; n = (n < Ncols) ? n : (Ncols - 1);
                f[j] = bp[n];
            }
        }
        v4u packed;
        packed.x = pack2bf(f[0], f[1]); packed.y = pack2bf(f[2], f[3]);
        packed.z = pack2bf(f[4], f[5]); packed.w = pack2bf(f[6], f[7]);
        *reinterpret_cast<v4u*>(lds_b + kb * 64 + nb) = packed;
    } else {
        // B^T: Bm is [Ncols,K]; lds_b[k][n] = Bm[col0+n, k0+k] (transpose)
        const int nb  = tid >> 2;          // 0..63
        const int kb8 = (tid & 3) << 3;    // 0,8,16,24
        int gn = col0 + nb; gn = (gn < Ncols) ? gn : (Ncols - 1);
        const float* bp = Bm + (size_t)gn * K + k0 + kb8;
        float f[8];
        if (FAST) {
            v4f a = *reinterpret_cast<const v4f*>(bp);
            v4f b = *reinterpret_cast<const v4f*>(bp + 4);
            f[0]=a.x; f[1]=a.y; f[2]=a.z; f[3]=a.w;
            f[4]=b.x; f[5]=b.y; f[6]=b.z; f[7]=b.w;
        } else {
            #pragma unroll
            for (int j = 0; j < 8; ++j) {
                const int gk = k0 + kb8 + j;
                f[j] = (gk < K) ? bp[j] : 0.f;
            }
        }
        #pragma unroll
        for (int j = 0; j < 8; ++j)
            lds_b[(kb8 + j) * 64 + nb] = f2bf_bits(f[j]);
    }
}

// ---------------------------------------------------------------------
// fragment build + 4 WMMAs for one 32-deep k-step
// A 16x32: lane&15 = M row; lanes<16 hold K 0-7 & 16-23, lanes>=16 K 8-15 & 24-31
// B 32x16: lane&15 = N col; lanes<16 hold K 0-7 & 16-23, lanes>=16 K 8-15 & 24-31
// ---------------------------------------------------------------------
static __device__ inline void compute_step(const unsigned short* __restrict__ la,
                                           const unsigned short* __restrict__ lb,
                                           int lane, int wr, int wc,
                                           v8f& a00, v8f& a01, v8f& a10, v8f& a11)
{
    const int ch = lane >> 4;              // 0 or 1
    Pair16 t;
    v16bf fa0, fa1, fb0, fb1;
    const v4u* pa0 = reinterpret_cast<const v4u*>(la + ((wr << 5) + (lane & 15)) * 32);
    t.lo = pa0[ch]; t.hi = pa0[ch + 2];
    fa0 = __builtin_bit_cast(v16bf, t);
    const v4u* pa1 = reinterpret_cast<const v4u*>(la + ((wr << 5) + 16 + (lane & 15)) * 32);
    t.lo = pa1[ch]; t.hi = pa1[ch + 2];
    fa1 = __builtin_bit_cast(v16bf, t);
    const int nb0 = (wc << 5) + (lane & 15);
    #pragma unroll
    for (int s = 0; s < 16; ++s) {
        const int kk = (s < 8) ? ((ch << 3) + s) : (16 + (ch << 3) + (s - 8));
        fb0[s] = __builtin_bit_cast(__bf16, lb[kk * 64 + nb0]);
        fb1[s] = __builtin_bit_cast(__bf16, lb[kk * 64 + nb0 + 16]);
    }

    a00 = __builtin_amdgcn_wmma_f32_16x16x32_bf16(false, fa0, false, fb0, (short)0, a00, false, false);
    a01 = __builtin_amdgcn_wmma_f32_16x16x32_bf16(false, fa0, false, fb1, (short)0, a01, false, false);
    a10 = __builtin_amdgcn_wmma_f32_16x16x32_bf16(false, fa1, false, fb0, (short)0, a10, false, false);
    a11 = __builtin_amdgcn_wmma_f32_16x16x32_bf16(false, fa1, false, fb1, (short)0, a11, false, false);
}

// =====================================================================
// Tiled GEMM:  C[M,Ncols] = epi( A[M,K] @ B + optional residual R )
// BT==0: B is [K,Ncols] row-major.  BT==1: B is [Ncols,K] row-major (B^T).
// EPI: 0 none, 1 relu, 2 add residual R[M,Ncols].
// Block = 256 threads (8 wave32). Block tile 128 rows x 64 cols.
// Waves 4x2; each wave owns a 32x32 C tile (4 x 16x16 f32 accums).
// Software-pipelined: ping-pong LDS buffers, ONE barrier per k-step,
// fast staging in the steady state, peeled guarded K-tail.
// Requires K >= 32 (all call sites have K in {384, 1152, 3600}).
// =====================================================================
template <int BT, int EPI>
__global__ __launch_bounds__(256) void gemm_wmma_kernel(
    float* __restrict__ C, const float* __restrict__ A,
    const float* __restrict__ Bm, const float* __restrict__ R,
    int M, int Ncols, int K)
{
    __shared__ unsigned short lds_a[2][128 * 32];   // 2 x 8 KB
    __shared__ unsigned short lds_b[2][32 * 64];    // 2 x 4 KB

    const int tid  = threadIdx.x;
    const int lane = tid & 31;
    const int w    = tid >> 5;
    const int wr   = w >> 1;        // 0..3 row tile (32 rows each)
    const int wc   = w & 1;         // 0..1 col tile (32 cols each)
    const int row0 = blockIdx.y * 128;
    const int col0 = blockIdx.x * 64;

    v8f a00 = {0.f,0.f,0.f,0.f,0.f,0.f,0.f,0.f};
    v8f a01 = a00, a10 = a00, a11 = a00;

    const int nFull = K >> 5;            // full 32-deep steps (>= 1)
    const bool tail = (K & 31) != 0;

    // prologue: stage step 0 into buffer 0
    stage_A<true>(lds_a[0], A, M, K, row0, 0, tid);
    stage_B<BT, true>(lds_b[0], Bm, Ncols, K, col0, 0, tid);
    __syncthreads();

    // steady state: stage step s while computing step s-1; one barrier/step
    for (int s = 1; s < nFull; ++s) {
        const int buf = s & 1;
        stage_A<true>(lds_a[buf], A, M, K, row0, s << 5, tid);
        stage_B<BT, true>(lds_b[buf], Bm, Ncols, K, col0, s << 5, tid);
        compute_step(lds_a[buf ^ 1], lds_b[buf ^ 1], lane, wr, wc, a00, a01, a10, a11);
        __syncthreads();
    }

    if (tail) {
        const int buf = nFull & 1;
        stage_A<false>(lds_a[buf], A, M, K, row0, nFull << 5, tid);
        stage_B<BT, false>(lds_b[buf], Bm, Ncols, K, col0, nFull << 5, tid);
        compute_step(lds_a[buf ^ 1], lds_b[buf ^ 1], lane, wr, wc, a00, a01, a10, a11);
        __syncthreads();
        compute_step(lds_a[buf], lds_b[buf], lane, wr, wc, a00, a01, a10, a11);
    } else {
        compute_step(lds_a[(nFull - 1) & 1], lds_b[(nFull - 1) & 1],
                     lane, wr, wc, a00, a01, a10, a11);
    }

    // ---- store C (VGPR j: lanes 0-15 -> M=j, lanes 16-31 -> M=8+j) ----
    const int mBase = row0 + (wr << 5) + ((lane >> 4) << 3);
    const int nBase = col0 + (wc << 5) + (lane & 15);
    #pragma unroll
    for (int r2 = 0; r2 < 2; ++r2) {
        #pragma unroll
        for (int c2 = 0; c2 < 2; ++c2) {
            const v8f& acc = (r2 == 0) ? (c2 == 0 ? a00 : a01)
                                       : (c2 == 0 ? a10 : a11);
            const int nn = nBase + (c2 << 4);
            #pragma unroll
            for (int j = 0; j < 8; ++j) {
                const int m = mBase + (r2 << 4) + j;
                if (m < M && nn < Ncols) {
                    float v = acc[j];
                    if (EPI == 1) v = v > 0.f ? v : 0.f;
                    if (EPI == 2) v += R[(size_t)m * Ncols + nn];
                    C[(size_t)m * Ncols + nn] = v;
                }
            }
        }
    }
}

// =====================================================================
// Row softmax over cols with x = s*scale + (mask? 0 : -10000)
// =====================================================================
__global__ void softmax_rows_kernel(float* __restrict__ S, int cols, float scale,
                                    const unsigned char* __restrict__ mask,
                                    int maskRowStride)
{
    __shared__ float red[256];
    const int row = blockIdx.x;
    const int tid = threadIdx.x;
    float* p = S + (size_t)row * cols;
    const unsigned char* mrow = mask + (size_t)row * maskRowStride;

    float mx = -3.4e38f;
    for (int c = tid; c < cols; c += 256) {
        float x = p[c] * scale + (mrow[c] ? 0.f : -10000.f);
        mx = fmaxf(mx, x);
    }
    red[tid] = mx; __syncthreads();
    for (int s = 128; s > 0; s >>= 1) {
        if (tid < s) red[tid] = fmaxf(red[tid], red[tid + s]);
        __syncthreads();
    }
    mx = red[0]; __syncthreads();

    float sum = 0.f;
    for (int c = tid; c < cols; c += 256) {
        float x = p[c] * scale + (mrow[c] ? 0.f : -10000.f);
        sum += __expf(x - mx);
    }
    red[tid] = sum; __syncthreads();
    for (int s = 128; s > 0; s >>= 1) {
        if (tid < s) red[tid] += red[tid + s];
        __syncthreads();
    }
    sum = red[0];
    const float inv = 1.f / sum;
    for (int c = tid; c < cols; c += 256) {
        float x = p[c] * scale + (mrow[c] ? 0.f : -10000.f);
        p[c] = __expf(x - mx) * inv;
    }
}

// =====================================================================
// LayerNorm over last dim (Dd), one block (128 thr) per row
// =====================================================================
__global__ void layernorm_kernel(float* __restrict__ out, const float* __restrict__ in,
                                 const float* __restrict__ g, const float* __restrict__ b,
                                 int Dd)
{
    __shared__ float r1[128];
    __shared__ float r2[128];
    const int row = blockIdx.x;
    const int tid = threadIdx.x;
    const float* x = in + (size_t)row * Dd;
    float s = 0.f, sq = 0.f;
    for (int c = tid; c < Dd; c += 128) { float v = x[c]; s += v; sq += v * v; }
    r1[tid] = s; r2[tid] = sq; __syncthreads();
    for (int st = 64; st > 0; st >>= 1) {
        if (tid < st) { r1[tid] += r1[tid + st]; r2[tid] += r2[tid + st]; }
        __syncthreads();
    }
    const float mean = r1[0] / (float)Dd;
    const float var  = r2[0] / (float)Dd - mean * mean;
    const float inv  = rsqrtf(var + LN_EPS);
    float* o = out + (size_t)row * Dd;
    for (int c = tid; c < Dd; c += 128) o[c] = (x[c] - mean) * inv * g[c] + b[c];
}

// =====================================================================
// out[idx] = dot(qv[b,:], kd[idx,:]) with b = idx / Nn  (idx over B*N)
// =====================================================================
__global__ void dot_rows_kernel(float* __restrict__ out, const float* __restrict__ qv,
                                const float* __restrict__ kd, int total, int Nn, int Dd)
{
    const int idx = blockIdx.x * blockDim.x + threadIdx.x;
    if (idx >= total) return;
    const int b = idx / Nn;
    const float* q = qv + (size_t)b * Dd;
    const float* k = kd + (size_t)idx * Dd;
    float acc = 0.f;
    for (int d = 0; d < Dd; ++d) acc += q[d] * k[d];
    out[idx] = acc;
}

// =====================================================================
// out[b,d] = sum_n p[b,n] * vd[b,n,d] ; grid = B blocks, blockDim = Dd
// =====================================================================
__global__ void wsum_kernel(float* __restrict__ out, const float* __restrict__ p,
                            const float* __restrict__ vd, int Nn, int Dd)
{
    const int b = blockIdx.x;
    const int d = threadIdx.x;
    if (d >= Dd) return;
    const float* pb = p + (size_t)b * Nn;
    const float* vb = vd + (size_t)b * Nn * Dd;
    float acc = 0.f;
    for (int n = 0; n < Nn; ++n) acc += pb[n] * vb[(size_t)n * Dd + d];
    out[(size_t)b * Dd + d] = acc;
}

// =====================================================================
// host orchestration
// =====================================================================
extern "C" void kernel_launch(void* const* d_in, const int* in_sizes, int n_in,
                              void* d_out, int out_size, void* d_ws, size_t ws_size,
                              hipStream_t stream)
{
    (void)in_sizes; (void)n_in; (void)out_size; (void)ws_size;

    const float* q_x   = (const float*)d_in[0];
    const float* s_x   = (const float*)d_in[1];
    const unsigned char* supp_mask = (const unsigned char*)d_in[2];
    const unsigned char* qry_mask  = (const unsigned char*)d_in[3];
    const float* proto_in = (const float*)d_in[4];
    const float* Wq  = (const float*)d_in[5];
    const float* Wk  = (const float*)d_in[6];
    const float* Wv  = (const float*)d_in[7];
    const float* Wp  = (const float*)d_in[8];
    const float* ln1g = (const float*)d_in[9];
    const float* ln1b = (const float*)d_in[10];
    const float* fw1 = (const float*)d_in[11];
    const float* fw2 = (const float*)d_in[12];
    const float* ln2g = (const float*)d_in[13];
    const float* ln2b = (const float*)d_in[14];
    const float* dWq = (const float*)d_in[15];
    const float* dWk = (const float*)d_in[16];
    const float* dWv = (const float*)d_in[17];
    const float* dWp = (const float*)d_in[18];
    const float* dng = (const float*)d_in[19];
    const float* dnb = (const float*)d_in[20];
    const float* mw1 = (const float*)d_in[21];
    const float* mw2 = (const float*)d_in[22];

    const size_t BND = (size_t)BATCH * NTOK * D_EMB;
    const size_t DD  = (size_t)D_EMB * D_EMB;
    const size_t DFF = (size_t)D_EMB * FF_DIM;

    float* ws = (float*)d_ws;
    size_t off = 0;
    auto carve = [&](size_t n) { float* p = ws + off; off += n; return p; };
    float* qbuf   = carve(BND);
    float* tmpq   = carve(BND);
    float* kbuf   = carve(BND);
    float* vbuf   = carve(BND);
    float* aobuf  = carve(BND);
    float* tmp2   = carve(BND);
    float* ffnmid = carve((size_t)BATCH * NTOK * FF_DIM);
    float* Sbuf   = carve((size_t)NTOK * NTOK);
    float* scores = carve((size_t)BATCH * NTOK);
    float* qp     = carve((size_t)BATCH * D_EMB);
    float* wsum   = carve((size_t)BATCH * D_EMB);
    float* protoA = carve((size_t)BATCH * D_EMB);
    float* protoB = carve((size_t)BATCH * D_EMB);
    float* pnbuf  = carve((size_t)BATCH * D_EMB);
    float* mmid   = carve((size_t)BATCH * D_EMB);
    float* mout   = carve((size_t)BATCH * D_EMB);

    auto gemm = [&](float* C, const float* A, const float* Bm, const float* R,
                    int M, int Ncols, int K, int bt, int epi) {
        dim3 grid((Ncols + 63) / 64, (M + 127) / 128);
        dim3 blk(256);
        if (bt == 1) {
            gemm_wmma_kernel<1, 0><<<grid, blk, 0, stream>>>(C, A, Bm, nullptr, M, Ncols, K);
            return;
        }
        switch (epi) {
        case 0: gemm_wmma_kernel<0, 0><<<grid, blk, 0, stream>>>(C, A, Bm, nullptr, M, Ncols, K); break;
        case 1: gemm_wmma_kernel<0, 1><<<grid, blk, 0, stream>>>(C, A, Bm, nullptr, M, Ncols, K); break;
        default: gemm_wmma_kernel<0, 2><<<grid, blk, 0, stream>>>(C, A, Bm, R, M, Ncols, K); break;
        }
    };

    hipMemcpyAsync(qbuf, q_x, BND * sizeof(float), hipMemcpyDeviceToDevice, stream);
    hipMemcpyAsync(protoA, proto_in, (size_t)BATCH * D_EMB * sizeof(float),
                   hipMemcpyDeviceToDevice, stream);

    float* pcur = protoA;
    float* pnxt = protoB;
    const int Mall = BATCH * NTOK;

    for (int l = 0; l < NLAYER; ++l) {
        const float* Wq_l = Wq + (size_t)l * DD;
        const float* Wk_l = Wk + (size_t)l * DD;
        const float* Wv_l = Wv + (size_t)l * DD;
        const float* Wp_l = Wp + (size_t)l * DD;
        const float* w1_l = fw1 + (size_t)l * DFF;
        const float* w2_l = fw2 + (size_t)l * DFF;
        const float* dWq_l = dWq + (size_t)l * DD;
        const float* dWk_l = dWk + (size_t)l * DD;
        const float* dWv_l = dWv + (size_t)l * DD;
        const float* dWp_l = dWp + (size_t)l * DD;

        // ---- q <- s cross attention ----
        gemm(tmpq, qbuf, Wq_l, nullptr, Mall, D_EMB, D_EMB, 0, 0);   // Q proj
        gemm(kbuf, s_x,  Wk_l, nullptr, Mall, D_EMB, D_EMB, 0, 0);   // K proj
        gemm(vbuf, s_x,  Wv_l, nullptr, Mall, D_EMB, D_EMB, 0, 0);   // V proj
        for (int b = 0; b < BATCH; ++b) {
            const size_t boff = (size_t)b * NTOK * D_EMB;
            // S = Q_b @ K_b^T  (S stays resident in 192MB L2)
            gemm(Sbuf, tmpq + boff, kbuf + boff, nullptr, NTOK, NTOK, D_EMB, 1, 0);
            softmax_rows_kernel<<<NTOK, 256, 0, stream>>>(
                Sbuf, NTOK, SCALE_F, supp_mask + (size_t)b * NTOK, 0);
            // O_b = P @ V_b
            gemm(aobuf + boff, Sbuf, vbuf + boff, nullptr, NTOK, D_EMB, NTOK, 0, 0);
        }
        gemm(tmp2, aobuf, Wp_l, qbuf, Mall, D_EMB, D_EMB, 0, 2);      // + residual q
        layernorm_kernel<<<Mall, 128, 0, stream>>>(qbuf, tmp2,
            ln1g + (size_t)l * D_EMB, ln1b + (size_t)l * D_EMB, D_EMB);

        // ---- FFN + LN ----
        gemm(ffnmid, qbuf, w1_l, nullptr, Mall, FF_DIM, D_EMB, 0, 1); // relu
        gemm(tmp2, ffnmid, w2_l, qbuf, Mall, D_EMB, FF_DIM, 0, 2);    // + residual q
        layernorm_kernel<<<Mall, 128, 0, stream>>>(qbuf, tmp2,
            ln2g + (size_t)l * D_EMB, ln2b + (size_t)l * D_EMB, D_EMB);

        // ---- proto <- q cross attention (1 query per batch) ----
        gemm(qp,   pcur, dWq_l, nullptr, BATCH, D_EMB, D_EMB, 0, 0);
        gemm(kbuf, qbuf, dWk_l, nullptr, Mall,  D_EMB, D_EMB, 0, 0);
        gemm(vbuf, qbuf, dWv_l, nullptr, Mall,  D_EMB, D_EMB, 0, 0);
        dot_rows_kernel<<<(Mall + 255) / 256, 256, 0, stream>>>(
            scores, qp, kbuf, Mall, NTOK, D_EMB);
        softmax_rows_kernel<<<BATCH, 256, 0, stream>>>(
            scores, NTOK, SCALE_F, qry_mask, NTOK);
        wsum_kernel<<<BATCH, D_EMB, 0, stream>>>(wsum, scores, vbuf, NTOK, D_EMB);
        gemm(pnxt, wsum, dWp_l, pcur, BATCH, D_EMB, D_EMB, 0, 2);     // + residual proto
        float* t = pcur; pcur = pnxt; pnxt = t;
    }

    // ---- mask head ----
    layernorm_kernel<<<BATCH, 128, 0, stream>>>(pnbuf, pcur, dng, dnb, D_EMB);
    gemm(mmid, pnbuf, mw1, nullptr, BATCH, D_EMB, D_EMB, 0, 1);       // relu
    gemm(mout, mmid,  mw2, pnbuf,  BATCH, D_EMB, D_EMB, 0, 2);        // + residual pn

    float* outq = (float*)d_out;
    hipMemcpyAsync(outq, qbuf, BND * sizeof(float), hipMemcpyDeviceToDevice, stream);
    dot_rows_kernel<<<(Mall + 255) / 256, 256, 0, stream>>>(
        outq + BND, mout, qbuf, Mall, NTOK, D_EMB);
}
